// CrossAttentionFusion_49297634623646
// MI455X (gfx1250) — compile-verified, matrix operand order checked by reading, not run on hardware
//
#include <hip/hip_runtime.h>

// ---- Problem constants (from reference) ----
#define EDIM   2048
#define MDIM   256
#define BATCH  16384
#define LN_EPS 1e-5f

typedef __bf16 bf16;
typedef __attribute__((ext_vector_type(16))) __bf16 v16bf;
typedef __attribute__((ext_vector_type(8)))  __bf16 v8bf;
typedef __attribute__((ext_vector_type(8)))  float  v8f;
typedef __attribute__((ext_vector_type(4)))  int    v4i;

// ---- CDNA5 async global->LDS path (gfx1250), with safe fallback ----
#if defined(__has_builtin)
#  if __has_builtin(__builtin_amdgcn_global_load_async_to_lds_b128)
#    define HAS_ASYNC_LDS 1
#  endif
#endif
#ifndef HAS_ASYNC_LDS
#  define HAS_ASYNC_LDS 0
#endif

#define LDS_AS __attribute__((address_space(3)))
#define GLB_AS __attribute__((address_space(1)))

__device__ __forceinline__ void wait_async_0() {
#if __has_builtin(__builtin_amdgcn_s_wait_asynccnt)
    __builtin_amdgcn_s_wait_asynccnt(0);
#else
    asm volatile("s_wait_asynccnt 0x0" ::: "memory");
#endif
}

#if HAS_ASYNC_LDS
__device__ __forceinline__ void async_cp16(const bf16* g, bf16* l) {
    // GLOBAL_LOAD_ASYNC_TO_LDS_B128: 16B per lane, tracked with ASYNCcnt.
    // Builtin signature (from hipcc diagnostic): (int4* __device__, int4* lds, imm, imm)
    __builtin_amdgcn_global_load_async_to_lds_b128(
        (GLB_AS v4i*)(g), (LDS_AS v4i*)(l), /*imm offset*/0, /*cpol*/0);
}
#endif

// ---------------------------------------------------------------------------
// fp32 -> bf16 conversion (vectorized: float4 in, 4x bf16 out)
// ---------------------------------------------------------------------------
__global__ void __launch_bounds__(256)
cvt_f32_bf16_v4(const float4* __restrict__ in, uint2* __restrict__ out, int n4) {
    int i = blockIdx.x * blockDim.x + threadIdx.x;
    int stride = gridDim.x * blockDim.x;
    for (; i < n4; i += stride) {
        float4 f = in[i];
        bf16 o[4];
        o[0] = (bf16)f.x; o[1] = (bf16)f.y; o[2] = (bf16)f.z; o[3] = (bf16)f.w;
        out[i] = *(const uint2*)o;
    }
}

// ---------------------------------------------------------------------------
// Tiled WMMA GEMM:  C[M,N] = A[M,K](bf16) * W[N,K]^T(bf16) + bias[N]
// 256 threads (8 waves) per block -> 128x128 C tile; wave grid 4x2, each wave
// computes 32x64 = 2x4 v_wmma_f32_16x16x32_bf16 fragments. Double-buffered
// LDS tiles filled by GLOBAL_LOAD_ASYNC_TO_LDS_B128 (ASYNCcnt pipelined).
// K and LDC are compile-time so store addressing folds into imm offsets.
// ---------------------------------------------------------------------------
#define LDS_STRIDE 40   // 32 + 8 pad: keeps 16B alignment, spreads LDS banks

template<bool OUT_BF16, int K, int LDC>
__global__ void __launch_bounds__(256)
gemm_bf16_wmma(const bf16* __restrict__ A, const bf16* __restrict__ W,
               const float* __restrict__ bias, void* __restrict__ Cout) {
    __shared__ alignas(16) bf16 As[2][128 * LDS_STRIDE];
    __shared__ alignas(16) bf16 Ws[2][128 * LDS_STRIDE];

    const int t    = threadIdx.x;
    const int lane = t & 31;
    const int wave = t >> 5;
    const int wm   = wave >> 1;        // 0..3 : 32-row band within 128
    const int wn   = wave & 1;         // 0..1 : 64-col band within 128
    const int half = lane >> 4;        // lane group 0/1 (K-offset select)
    const int l16  = lane & 15;

    const int bm = blockIdx.x;         // M / 128
    const int bn = blockIdx.y;         // N / 128

    v8f acc[2][4];
    const v8f vzero = {0.f, 0.f, 0.f, 0.f, 0.f, 0.f, 0.f, 0.f};
#pragma unroll
    for (int i = 0; i < 2; ++i)
#pragma unroll
        for (int j = 0; j < 4; ++j) acc[i][j] = vzero;

    // global->LDS copy mapping: thread t moves 16B at (row = t>>2 (+64), col8 = (t&3)*8)
    const int r0 = t >> 2;
    const int c8 = (t & 3) * 8;
    const bf16* gA = A + (size_t)(bm * 128 + r0) * K + c8;
    const bf16* gW = W + (size_t)(bn * 128 + r0) * K + c8;

    constexpr int NK = K / 32;

    // ---- fragment compute on one LDS buffer pair (8 WMMAs) ----
    auto compute = [&](const bf16* as, const bf16* ws) {
        // CDNA5 16-bit 16x32 operand layout (ISA 7.12.2): lanes 0-15 hold
        // K 0-7 / 16-23, lanes 16-31 hold K 8-15 / 24-31; two ds_load_b128.
        v16bf afrag[2], bfrag[4];
#pragma unroll
        for (int i = 0; i < 2; ++i) {
            const bf16* p = &as[(wm * 32 + i * 16 + l16) * LDS_STRIDE + half * 8];
            v8bf lo = *(const v8bf*)(p);
            v8bf hi = *(const v8bf*)(p + 16);
            afrag[i] = __builtin_shufflevector(lo, hi, 0,1,2,3,4,5,6,7,8,9,10,11,12,13,14,15);
        }
#pragma unroll
        for (int j = 0; j < 4; ++j) {
            const bf16* p = &ws[(wn * 64 + j * 16 + l16) * LDS_STRIDE + half * 8];
            v8bf lo = *(const v8bf*)(p);
            v8bf hi = *(const v8bf*)(p + 16);
            bfrag[j] = __builtin_shufflevector(lo, hi, 0,1,2,3,4,5,6,7,8,9,10,11,12,13,14,15);
        }
#pragma unroll
        for (int i = 0; i < 2; ++i)
#pragma unroll
            for (int j = 0; j < 4; ++j)
                acc[i][j] = __builtin_amdgcn_wmma_f32_16x16x32_bf16(
                    false, afrag[i], false, bfrag[j],
                    (short)0, acc[i][j], false, false);
    };

#if HAS_ASYNC_LDS
    // ---- double-buffered pipeline over ASYNCcnt ----
    auto issue = [&](int buf) {
        bf16* la = &As[buf][r0 * LDS_STRIDE + c8];
        bf16* lw = &Ws[buf][r0 * LDS_STRIDE + c8];
        async_cp16(gA,          la);
        async_cp16(gA + 64 * K, la + 64 * LDS_STRIDE);
        async_cp16(gW,          lw);
        async_cp16(gW + 64 * K, lw + 64 * LDS_STRIDE);
        gA += 32; gW += 32;
    };
    issue(0);
    int buf = 0;
    for (int kt = 0; kt < NK; ++kt) {
        wait_async_0();        // this wave's tile(kt) loads done
        __syncthreads();       // => all waves' loads done AND all waves
                               //    finished reading the other buffer
        if (kt + 1 < NK) issue(buf ^ 1);
        compute(&As[buf][0], &Ws[buf][0]);
        buf ^= 1;
    }
#else
    // ---- synchronous fallback: VGPR-staged copy, single buffer ----
    for (int kt = 0; kt < NK; ++kt) {
        __syncthreads();       // previous iteration's frag reads done
        uint4 va0 = *(const uint4*)(gA);
        uint4 va1 = *(const uint4*)(gA + 64 * K);
        uint4 vw0 = *(const uint4*)(gW);
        uint4 vw1 = *(const uint4*)(gW + 64 * K);
        *(uint4*)&As[0][r0        * LDS_STRIDE + c8] = va0;
        *(uint4*)&As[0][(r0 + 64) * LDS_STRIDE + c8] = va1;
        *(uint4*)&Ws[0][r0        * LDS_STRIDE + c8] = vw0;
        *(uint4*)&Ws[0][(r0 + 64) * LDS_STRIDE + c8] = vw1;
        gA += 32; gW += 32;
        __syncthreads();
        compute(&As[0][0], &Ws[0][0]);
    }
#endif

    // ---- epilogue; C/D layout: lane holds N = l16, M = v + 8*half ----
#pragma unroll
    for (int j = 0; j < 4; ++j) {
        const int col = bn * 128 + wn * 64 + j * 16 + l16;
        const float bv = bias[col];
#pragma unroll
        for (int i = 0; i < 2; ++i) {
            const int rbase = bm * 128 + wm * 32 + i * 16 + half * 8;
            if (OUT_BF16) {
                bf16* p = (bf16*)Cout + (size_t)rbase * LDC + col;
#pragma unroll
                for (int v = 0; v < 8; ++v) p[v * LDC] = (bf16)(acc[i][j][v] + bv);
            } else {
                float* p = (float*)Cout + (size_t)rbase * LDC + col;
#pragma unroll
                for (int v = 0; v < 8; ++v) p[v * LDC] = acc[i][j][v] + bv;
            }
        }
    }
}

// ---------------------------------------------------------------------------
// LayerNorm(attended + image) with concat output.
// attended pre-stored in out[:, E:2E]; normalize in place, copy image to
// out[:, 0:E]. One 256-thread block per row.
// ---------------------------------------------------------------------------
__global__ void __launch_bounds__(256)
ln_concat(const float* __restrict__ img,
          const float* __restrict__ gamma, const float* __restrict__ beta,
          float* __restrict__ out) {
    const int row = blockIdx.x;
    const int t   = threadIdx.x;
    __shared__ float red[16];

    const float4* irow = (const float4*)(img + (size_t)row * EDIM);
    float4* out0 = (float4*)(out + (size_t)row * (2 * EDIM));
    float4* out1 = (float4*)(out + (size_t)row * (2 * EDIM) + EDIM);

    float x[8];
    float s = 0.f, ss = 0.f;
#pragma unroll
    for (int j = 0; j < 2; ++j) {
        const int idx = t + j * 256;
        float4 a  = out1[idx];              // attended (written by GEMM3)
        float4 im = irow[idx];
        out0[idx] = im;                     // concat: first half = image
        float4 v;
        v.x = a.x + im.x; v.y = a.y + im.y; v.z = a.z + im.z; v.w = a.w + im.w;
        x[j*4+0] = v.x; x[j*4+1] = v.y; x[j*4+2] = v.z; x[j*4+3] = v.w;
        s  += v.x + v.y + v.z + v.w;
        ss += v.x*v.x + v.y*v.y + v.z*v.z + v.w*v.w;
    }
#pragma unroll
    for (int off = 16; off >= 1; off >>= 1) {
        s  += __shfl_xor(s,  off, 32);
        ss += __shfl_xor(ss, off, 32);
    }
    const int wv = t >> 5, lane = t & 31;
    if (lane == 0) { red[wv] = s; red[8 + wv] = ss; }
    __syncthreads();
    s = 0.f; ss = 0.f;
#pragma unroll
    for (int w = 0; w < 8; ++w) { s += red[w]; ss += red[8 + w]; }

    const float inv_n = 1.0f / (float)EDIM;
    const float mean  = s * inv_n;
    const float var   = ss * inv_n - mean * mean;
    const float rstd  = rsqrtf(var + LN_EPS);

    const float4* g4 = (const float4*)gamma;
    const float4* b4 = (const float4*)beta;
#pragma unroll
    for (int j = 0; j < 2; ++j) {
        const int idx = t + j * 256;
        float4 g = g4[idx], b = b4[idx];
        float4 o;
        o.x = (x[j*4+0] - mean) * rstd * g.x + b.x;
        o.y = (x[j*4+1] - mean) * rstd * g.y + b.y;
        o.z = (x[j*4+2] - mean) * rstd * g.z + b.z;
        o.w = (x[j*4+3] - mean) * rstd * g.w + b.w;
        out1[idx] = o;
    }
}

// ---------------------------------------------------------------------------
// softmax over a single key == 1  =>  attended == v: Q and K GEMMs are dead.
//   cvt:   meta, W_meta, Wv, Wout -> bf16 (workspace)
//   GEMM1: meta_kv = meta @ W_meta^T + b_meta            (bf16 out)
//   GEMM2: v       = meta_kv @ Wv^T + bv                 (bf16 out)
//   GEMM3: att     = v @ out_proj^T + b_out -> out[:, E:2E] (fp32, ldc=2E)
//   LN:    out[:, E:2E] = LN(att + image); out[:, 0:E] = image
// ---------------------------------------------------------------------------
extern "C" void kernel_launch(void* const* d_in, const int* in_sizes, int n_in,
                              void* d_out, int out_size, void* d_ws, size_t ws_size,
                              hipStream_t stream) {
    const float* img   = (const float*)d_in[0];
    const float* meta  = (const float*)d_in[1];
    const float* Wmeta = (const float*)d_in[2];
    const float* bmeta = (const float*)d_in[3];
    const float* inW   = (const float*)d_in[4];
    const float* inB   = (const float*)d_in[5];
    const float* Wout  = (const float*)d_in[6];
    const float* bout  = (const float*)d_in[7];
    const float* gamma = (const float*)d_in[8];
    const float* beta  = (const float*)d_in[9];

    const float* Wv = inW + (size_t)2 * EDIM * EDIM;   // V rows of packed in-proj
    const float* bv = inB + (size_t)2 * EDIM;

    char* ws = (char*)d_ws;
    size_t off = 0;
    auto carve = [&](size_t bytes) {
        void* p = ws + off;
        off += (bytes + 255) & ~(size_t)255;
        return p;
    };
    bf16* meta_h  = (bf16*)carve((size_t)BATCH * MDIM * sizeof(bf16));
    bf16* Wmeta_h = (bf16*)carve((size_t)EDIM * MDIM * sizeof(bf16));
    bf16* Wv_h    = (bf16*)carve((size_t)EDIM * EDIM * sizeof(bf16));
    bf16* Wo_h    = (bf16*)carve((size_t)EDIM * EDIM * sizeof(bf16));
    bf16* kv_h    = (bf16*)carve((size_t)BATCH * EDIM * sizeof(bf16));
    bf16* v_h     = (bf16*)carve((size_t)BATCH * EDIM * sizeof(bf16));

    float* outp = (float*)d_out;

    auto cvt = [&](const float* src, bf16* dst, size_t n) {
        int n4 = (int)(n / 4);
        int blocks = (n4 + 255) / 256;
        if (blocks > 16384) blocks = 16384;
        cvt_f32_bf16_v4<<<blocks, 256, 0, stream>>>((const float4*)src, (uint2*)dst, n4);
    };
    cvt(meta,  meta_h,  (size_t)BATCH * MDIM);
    cvt(Wmeta, Wmeta_h, (size_t)EDIM * MDIM);
    cvt(Wv,    Wv_h,    (size_t)EDIM * EDIM);
    cvt(Wout,  Wo_h,    (size_t)EDIM * EDIM);

    dim3 grid(BATCH / 128, EDIM / 128);
    gemm_bf16_wmma<true,  MDIM, EDIM    ><<<grid, 256, 0, stream>>>(meta_h, Wmeta_h, bmeta, kv_h);
    gemm_bf16_wmma<true,  EDIM, EDIM    ><<<grid, 256, 0, stream>>>(kv_h,   Wv_h,    bv,    v_h);
    gemm_bf16_wmma<false, EDIM, 2*EDIM  ><<<grid, 256, 0, stream>>>(v_h,    Wo_h,    bout,  outp + EDIM);
    ln_concat<<<BATCH, 256, 0, stream>>>(img, gamma, beta, outp);
}
